// RestorationModel_163208757641
// MI455X (gfx1250) — matrix operand order repaired and checked by reading, not compiled, not run on hardware
//
#include <hip/hip_runtime.h>

#define FEAT 2048
#define HID  1024
#define BB   8
#define TT   1024
#define MPAD (TT * 16)   // 16384 padded rows (t*16 + b)

typedef _Float16 v16h __attribute__((ext_vector_type(16)));
typedef _Float16 v8h  __attribute__((ext_vector_type(8)));
typedef float    v8f  __attribute__((ext_vector_type(8)));

static __device__ __forceinline__ v8f wmma16(v16h a, v16h b, v8f c) {
  // D = A(16x32 f16) * B(32x16 f16) + C(16x16 f32)
  return __builtin_amdgcn_wmma_f32_16x16x32_f16(false, a, false, b, (short)0, c,
                                                false, false);
}
static __device__ __forceinline__ v16h cat8(v8h lo, v8h hi) {
  return __builtin_shufflevector(lo, hi, 0, 1, 2, 3, 4, 5, 6, 7, 8, 9, 10, 11,
                                 12, 13, 14, 15);
}

// ---------------------------------------------------------------- utilities
__global__ __launch_bounds__(256) void k_init(unsigned* bar, _Float16* zrow,
                                              int zn) {
  if (blockIdx.x == 0 && threadIdx.x < 2) bar[threadIdx.x] = 0u;
  int i = blockIdx.x * blockDim.x + threadIdx.x;
  if (i < zn) zrow[i] = (_Float16)0.f;
}

__global__ __launch_bounds__(256) void k_f32_to_f16(const float* __restrict__ s,
                                                    _Float16* __restrict__ d,
                                                    long n) {
  long i = (long)blockIdx.x * blockDim.x + threadIdx.x;
  if (i < n) d[i] = (_Float16)s[i];
}

// W[k][n] (row-major KxN, f32) -> Wt[n][k] (col-major: N rows of K contiguous, f16)
__global__ __launch_bounds__(256) void k_trcvt(const float* __restrict__ W,
                                               _Float16* __restrict__ Wt,
                                               int K, int N) {
  long i = (long)blockIdx.x * blockDim.x + threadIdx.x;
  if (i >= (long)K * N) return;
  int k = (int)(i / N), n = (int)(i % N);
  Wt[(long)n * K + k] = (_Float16)W[i];
}

// ---------------------------------------------------------------- big GEMM
// A: f16 row-major [*, K]. Bt: f16 col-major [N][K]. bias: f32[N].
// Out rows live in padded space r = t*16 + b (M = MPAD).
// AMAP: 0 = A row == r (A is [MPAD][K]);
//       1 = A row = b*TT + t; pad lanes (b>=8) read a zeroed dummy row.
// OMAP: 0 = C[r][N] f32 (padded layout);
//       1 = write only b<8 rows to C[(b*TT+t)][N] (d_out layout).
template <int AMAP, int OMAP>
__global__ __launch_bounds__(256) void k_gemm(const _Float16* __restrict__ A,
                                              const _Float16* __restrict__ Bt,
                                              const float* __restrict__ bias,
                                              float* __restrict__ C,
                                              const _Float16* __restrict__ zrow,
                                              int N, int K) {
  const int lane = threadIdx.x & 31;
  const int w    = threadIdx.x >> 5;
  const int wm   = w & 1;        // 2 M-waves
  const int wn   = w >> 1;       // 4 N-waves
  const int m0   = blockIdx.y * 128 + wm * 64;  // wave M base: 4 subtiles of 16
  const int n0   = blockIdx.x * 128 + wn * 32;  // wave N base: 2 subtiles of 16
  const int mrow = lane & 15;
  const int hi   = lane >> 4;    // K-half select (contiguous k = 16*hi + s for A and B)

  // Hoist per-fragment A base pointers; invalid pad lanes point at a zero row
  // so the inner loop is branch-free (EXEC stays all-1s around every WMMA).
  const _Float16* abase[4];
#pragma unroll
  for (int mi = 0; mi < 4; ++mi) {
    if (AMAP == 0) {
      abase[mi] = A + (long)(m0 + mi * 16 + mrow) * K;
    } else {
      int t = m0 / 16 + mi;
      abase[mi] = (mrow < BB) ? (A + ((long)mrow * TT + t) * K) : zrow;
    }
  }
  const _Float16* bbase[2];
#pragma unroll
  for (int ni = 0; ni < 2; ++ni)
    bbase[ni] = Bt + (long)(n0 + ni * 16 + mrow) * K + hi * 16;

  v8f acc[4][2] = {};
  for (int k0 = 0; k0 < K; k0 += 32) {
    v16h afr[4];
#pragma unroll
    for (int mi = 0; mi < 4; ++mi) {
      const _Float16* p = abase[mi] + k0 + hi * 16;
      afr[mi] = *(const v16h*)p;
    }
    v16h bfr[2];
#pragma unroll
    for (int ni = 0; ni < 2; ++ni)
      bfr[ni] = *(const v16h*)(bbase[ni] + k0);
#pragma unroll
    for (int mi = 0; mi < 4; ++mi)
#pragma unroll
      for (int ni = 0; ni < 2; ++ni)
        acc[mi][ni] = wmma16(afr[mi], bfr[ni], acc[mi][ni]);
  }

  // epilogue: C/D layout -> lane holds N = n0+ni*16+mrow, rows r + hi*8
#pragma unroll
  for (int mi = 0; mi < 4; ++mi)
#pragma unroll
    for (int ni = 0; ni < 2; ++ni) {
      int   n  = n0 + ni * 16 + mrow;
      float bv = bias[n];
      if (OMAP == 0) {
        int mb = m0 + mi * 16 + hi * 8;
#pragma unroll
        for (int r = 0; r < 8; ++r)
          C[(long)(mb + r) * N + n] = acc[mi][ni][r] + bv;
      } else {
        // out row = b*TT + t with b = hi*8 + r; only b<8 (hi==0) is real data
        if (hi == 0) {
          int t = m0 / 16 + mi;
#pragma unroll
          for (int r = 0; r < 8; ++r)
            C[((long)r * TT + t) * (long)N + n] = acc[mi][ni][r] + bv;
        }
      }
    }
}

// ---------------------------------------------------------------- recurrence
static __device__ __forceinline__ void grid_barrier(unsigned* cnt, unsigned* gen,
                                                    unsigned nwg) {
  __threadfence();   // release: every wave drains its own global stores
  __syncthreads();
  if (threadIdx.x == 0) {
    unsigned g = __hip_atomic_load(gen, __ATOMIC_RELAXED, __HIP_MEMORY_SCOPE_AGENT);
    if (__hip_atomic_fetch_add(cnt, 1u, __ATOMIC_ACQ_REL,
                               __HIP_MEMORY_SCOPE_AGENT) == nwg - 1) {
      __hip_atomic_store(cnt, 0u, __ATOMIC_RELAXED, __HIP_MEMORY_SCOPE_AGENT);
      __hip_atomic_fetch_add(gen, 1u, __ATOMIC_RELEASE, __HIP_MEMORY_SCOPE_AGENT);
    } else {
      while (__hip_atomic_load(gen, __ATOMIC_ACQUIRE,
                               __HIP_MEMORY_SCOPE_AGENT) == g)
        __builtin_amdgcn_s_sleep(1);
    }
  }
  __syncthreads();
  __threadfence();   // acquire: invalidate stale cached lines of peer state
}

// Persistent recurrence: h_t = relu(xw_t + h_{t-1} @ U), 16 rows (8 valid).
// XW: f32 [T*16][HID] padded layout. Ut: f16 col-major [HID][HID].
// Hseq: f16 [T*16][HID] output (also the h state carried between steps).
// 16 workgroups x 256 threads; WG owns 64 hidden columns; its 128KB U tile is
// staged once into LDS (320KB/WGP on CDNA5) in WMMA B-fragment order.
#define RNN_NWG 16
#define RNN_UFRAGS (4 * 32)                       // 4 col-tiles x 32 k-steps
#define RNN_SMEM_BYTES (RNN_UFRAGS * 32 * 32 + 8 * 32 * 32)  // 131072 + 8192

__global__ __launch_bounds__(256) void k_rnn(const float* __restrict__ XW,
                                             const _Float16* __restrict__ Ut,
                                             _Float16* __restrict__ Hseq,
                                             unsigned* __restrict__ bar,
                                             int T) {
  extern __shared__ v16h smem_v[];                 // 32B aligned
  v16h* uLds = smem_v;                             // [nt*32+ks][lane] fragments
  v8f*  redv = (v8f*)(smem_v + RNN_UFRAGS * 32);   // [wave][lane] partials

  const int lane = threadIdx.x & 31;
  const int w    = threadIdx.x >> 5;
  const int wgN  = blockIdx.x * 64;
  const int mrow = lane & 15;
  const int hi   = lane >> 4;

  // Stage U tile into LDS in fragment-linear order (one v16h per lane/frag).
  for (int f = w; f < RNN_UFRAGS; f += 8) {
    int nt  = f >> 5, ks = f & 31;
    int col = wgN + nt * 16 + mrow;
    int k0  = ks * 32 + hi * 16;
    uLds[f * 32 + lane] = *(const v16h*)(Ut + (long)col * HID + k0);
  }
  __syncthreads();

  const int nt      = w & 3;   // wave's column tile (16 cols)
  const int kh      = w >> 2;  // wave's K half (512 each)
  const int colBase = wgN + nt * 16;

  for (int t = 0; t < T; ++t) {
    v8f c = {};
    if (t > 0) {
      const _Float16* hprev = Hseq + (long)(t - 1) * 16 * HID;
#pragma unroll 4
      for (int ks = 0; ks < 16; ++ks) {
        int kstep = kh * 16 + ks;
        int kb    = kstep * 32 + hi * 16;
        const _Float16* pa = hprev + mrow * HID + kb;
        v16h a = *(const v16h*)pa;
        v16h b = uLds[(nt * 32 + kstep) * 32 + lane];
        c = wmma16(a, b, c);
      }
    }
    redv[w * 32 + lane] = c;
    __syncthreads();
    if (w < 4) {  // combine K halves, add xw, relu, emit f16 state
      v8f s = redv[w * 32 + lane] + redv[(w + 4) * 32 + lane];
      int n  = colBase + mrow;
      int mo = hi * 8;
      const float* xwt = XW + (long)t * 16 * HID;
      _Float16* hout   = Hseq + (long)t * 16 * HID;
#pragma unroll
      for (int r = 0; r < 8; ++r) {
        float v = s[r] + xwt[(mo + r) * HID + n];
        hout[(mo + r) * HID + n] = (_Float16)(v > 0.f ? v : 0.f);
      }
    }
    __syncthreads();  // redv reuse safety
    grid_barrier(bar, bar + 1, RNN_NWG);
  }
}

// ---------------------------------------------------------------- launcher
extern "C" void kernel_launch(void* const* d_in, const int* in_sizes, int n_in,
                              void* d_out, int out_size, void* d_ws,
                              size_t ws_size, hipStream_t stream) {
  const float* x  = (const float*)d_in[0];
  const float* W1 = (const float*)d_in[1];
  const float* U1 = (const float*)d_in[2];
  const float* b1 = (const float*)d_in[3];
  const float* W2 = (const float*)d_in[4];
  const float* U2 = (const float*)d_in[5];
  const float* b2 = (const float*)d_in[6];
  const float* Wp = (const float*)d_in[7];
  const float* bp = (const float*)d_in[8];
  const float* Wn = (const float*)d_in[9];
  const float* bn = (const float*)d_in[10];
  float* out = (float*)d_out;

  char* ws = (char*)d_ws;
  size_t off = 0;
  auto take = [&](size_t bytes) {
    char* p = ws + off;
    off += (bytes + 255) & ~(size_t)255;
    return p;
  };
  unsigned*  bar  = (unsigned*)take(256);
  _Float16*  zrow = (_Float16*)take((size_t)FEAT * 2);            // zero A row
  _Float16*  xf16 = (_Float16*)take((size_t)BB * TT * FEAT * 2);  // 32 MB
  _Float16*  w1t  = (_Float16*)take((size_t)HID * FEAT * 2);      // [1024][2048]
  _Float16*  u1t  = (_Float16*)take((size_t)HID * HID * 2);
  _Float16*  w2t  = (_Float16*)take((size_t)HID * HID * 2);
  _Float16*  u2t  = (_Float16*)take((size_t)HID * HID * 2);
  _Float16*  wpt  = (_Float16*)take((size_t)FEAT * HID * 2);      // [2048][1024]
  _Float16*  wnt  = (_Float16*)take((size_t)FEAT * HID * 2);
  float*     xw   = (float*)take((size_t)MPAD * HID * 4);         // 64 MB, reused
  _Float16*  h1   = (_Float16*)take((size_t)MPAD * HID * 2);      // 32 MB
  _Float16*  h2   = (_Float16*)take((size_t)MPAD * HID * 2);      // 32 MB
  (void)ws_size; (void)n_in; (void)in_sizes; (void)out_size;

  (void)hipFuncSetAttribute((const void*)k_rnn,
                            hipFuncAttributeMaxDynamicSharedMemorySize,
                            RNN_SMEM_BYTES);

  // 0) barrier + zero-row init (deterministic per call)
  k_init<<<(FEAT + 255) / 256, 256, 0, stream>>>(bar, zrow, FEAT);

  // 1) precision/layout conversions
  {
    long n = (long)BB * TT * FEAT;
    k_f32_to_f16<<<(unsigned)((n + 255) / 256), 256, 0, stream>>>(x, xf16, n);
  }
  k_trcvt<<<(FEAT * HID + 255) / 256, 256, 0, stream>>>(W1, w1t, FEAT, HID);
  k_trcvt<<<(HID * HID + 255) / 256, 256, 0, stream>>>(U1, u1t, HID, HID);
  k_trcvt<<<(HID * HID + 255) / 256, 256, 0, stream>>>(W2, w2t, HID, HID);
  k_trcvt<<<(HID * HID + 255) / 256, 256, 0, stream>>>(U2, u2t, HID, HID);
  k_trcvt<<<(HID * FEAT + 255) / 256, 256, 0, stream>>>(Wp, wpt, HID, FEAT);
  k_trcvt<<<(HID * FEAT + 255) / 256, 256, 0, stream>>>(Wn, wnt, HID, FEAT);

  // 2) xw1 = x @ W1 + b1  (padded [t*16+b] layout)
  k_gemm<1, 0><<<dim3(HID / 128, MPAD / 128), 256, 0, stream>>>(
      xf16, w1t, b1, xw, zrow, HID, FEAT);

  // 3) rnn1 recurrence -> h1
  k_rnn<<<RNN_NWG, 256, RNN_SMEM_BYTES, stream>>>(xw, u1t, h1, bar, TT);

  // 4) xw2 = h1 @ W2 + b2
  k_gemm<0, 0><<<dim3(HID / 128, MPAD / 128), 256, 0, stream>>>(
      h1, w2t, b2, xw, zrow, HID, HID);

  // 5) rnn2 recurrence -> h2
  k_rnn<<<RNN_NWG, 256, RNN_SMEM_BYTES, stream>>>(xw, u2t, h2, bar, TT);

  // 6) piano = h2 @ Wp + bp ; noise = h2 @ Wn + bn  (direct to d_out)
  k_gemm<0, 1><<<dim3(FEAT / 128, MPAD / 128), 256, 0, stream>>>(
      h2, wpt, bp, out, zrow, FEAT, HID);
  k_gemm<0, 1><<<dim3(FEAT / 128, MPAD / 128), 256, 0, stream>>>(
      h2, wnt, bn, out + (size_t)BB * TT * FEAT, zrow, FEAT, HID);
}
